// MoELayer_24584392802845
// MI455X (gfx1250) — compile-verified, matrix operand order checked by reading, not run on hardware
//
#include <hip/hip_runtime.h>
#include <hip/hip_bf16.h>
#include <math.h>

// ---------------- problem constants ----------------
static constexpr int Bb = 2, Ss = 2048, Hh = 1024, Ee = 8, Kk = 2;
static constexpr int FFf = 4 * Hh;          // 4096
static constexpr int TOK = Bb * Ss;         // 4096 tokens
static constexpr float EPS = 1e-5f;
static constexpr int BM = 32;               // tokens per FFN workgroup
static constexpr int MAX_TILES = TOK / BM;  // 128 (worst case: all tokens -> one expert)

// ---------------- workspace layout (bytes) ----------------
static constexpr size_t OFF_W1T = 0;                                    // bf16 [E][FF][H]
static constexpr size_t SZ_W1T  = (size_t)Ee * FFf * Hh * 2;            // 64 MB
static constexpr size_t OFF_W2T = OFF_W1T + SZ_W1T;                     // bf16 [E][H][FF]
static constexpr size_t SZ_W2T  = (size_t)Ee * Hh * FFf * 2;            // 64 MB
static constexpr size_t OFF_YB  = OFF_W2T + SZ_W2T;                     // f32 [2][TOK][H]
static constexpr size_t SZ_YB   = (size_t)2 * TOK * Hh * 4;             // 32 MB
static constexpr size_t OFF_MU  = OFF_YB + SZ_YB;                       // f32 [TOK]
static constexpr size_t OFF_RS  = OFF_MU + (size_t)TOK * 4;             // f32 [TOK]
static constexpr size_t OFF_CNT = OFF_RS + (size_t)TOK * 4;             // i32 [E] (pad 128)
static constexpr size_t OFF_PRM = OFF_CNT + 128;                        // i32 [E][TOK]  (token*2+slot)
static constexpr size_t OFF_GTW = OFF_PRM + (size_t)Ee * TOK * 4;       // f32 [E][TOK]

// ---------------- CDNA5 async global->LDS path (probe via __has_builtin) ----------------
#if defined(__HIP_DEVICE_COMPILE__) && __has_builtin(__builtin_amdgcn_global_load_async_to_lds_b128)
  #define HAVE_ASYNC_LDS 1
#else
  #define HAVE_ASYNC_LDS 0
#endif

typedef int v4i __attribute__((ext_vector_type(4)));

#if HAVE_ASYNC_LDS
typedef __attribute__((address_space(1))) v4i* gv4i_p;   // global int4*
typedef __attribute__((address_space(3))) v4i* lv4i_p;   // LDS int4*
  #if __has_builtin(__builtin_amdgcn_s_wait_asynccnt)
    #define WAIT_STAGE(n) __builtin_amdgcn_s_wait_asynccnt(n)
  #else
    #define WAIT_STAGE(n) asm volatile("s_wait_asynccnt %0" ::"n"(n) : "memory")
  #endif
#else
  #define WAIT_STAGE(n) ((void)0)
#endif

// ---------------- WMMA types ----------------
typedef __attribute__((ext_vector_type(16))) __bf16 v16bf;
typedef __attribute__((ext_vector_type(8)))  float  v8f;

union FragU { uint4 q[2]; v16bf v; };

// Load a bf16 16xK-slice fragment (A or B operand) from LDS.
// base_row must already include  (tile_row + (lane&15))*stride + 8*(lane>>4).
// Layout per CDNA5 ISA 7.12.2 (16-bit A 16x32): lanes 0-15 hold K {k0..k0+7, k0+16..k0+23},
// lanes 16-31 the +8 shifted halves -> two contiguous 16-byte LDS reads.
__device__ __forceinline__ v16bf load_frag(const __bf16* base_row, int k0) {
    FragU u;
    u.q[0] = *(const uint4*)(base_row + k0);
    u.q[1] = *(const uint4*)(base_row + k0 + 16);
    return u.v;
}

// Stage a 64(row) x 64(col) bf16 tile from global into LDS (256 threads, 32B/thread).
// Async path: GLOBAL_LOAD_ASYNC_TO_LDS_B128 (ASYNCcnt); fallback: b128 via VGPRs.
__device__ __forceinline__ void stage64x64(const __bf16* __restrict__ src,
                                           size_t rowStrideElems,
                                           __bf16* __restrict__ dst, int tid) {
    const int row = tid >> 2, seg = tid & 3;
    const __bf16* s = src + (size_t)row * rowStrideElems + seg * 16;
    __bf16* d = dst + row * 64 + seg * 16;
#if HAVE_ASYNC_LDS
    __builtin_amdgcn_global_load_async_to_lds_b128(
        (gv4i_p)(const void*)s, (lv4i_p)(void*)d, 0, 0);
    __builtin_amdgcn_global_load_async_to_lds_b128(
        (gv4i_p)(const void*)(s + 8), (lv4i_p)(void*)(d + 8), 0, 0);
#else
    const uint4* sv = (const uint4*)s;
    uint4* dv = (uint4*)d;
    dv[0] = sv[0];
    dv[1] = sv[1];
#endif
}

// ---------------- kernel 1: fp32 -> bf16 transpose-convert (per expert) ----------------
// src: fp32 [E][R][C]  ->  dst: bf16 [E][C][R]
__global__ void transpose_cvt_kernel(const float* __restrict__ src, __bf16* __restrict__ dst,
                                     int R, int C) {
    __shared__ float tile[32][33];
    const int e  = blockIdx.z;
    const int c0 = blockIdx.x * 32;
    const int r0 = blockIdx.y * 32;
    const float* s = src + (size_t)e * R * C;
    __bf16* d = dst + (size_t)e * R * C;
#pragma unroll
    for (int j = 0; j < 4; ++j) {
        int r = r0 + threadIdx.y + j * 8;
        tile[threadIdx.y + j * 8][threadIdx.x] = s[(size_t)r * C + c0 + threadIdx.x];
    }
    __syncthreads();
#pragma unroll
    for (int j = 0; j < 4; ++j) {
        int c = c0 + threadIdx.y + j * 8;
        d[(size_t)c * R + r0 + threadIdx.x] = (__bf16)tile[threadIdx.x][threadIdx.y + j * 8];
    }
}

// ---------------- kernel 2: router + LN stats + bucket compaction ----------------
// One wave32 per token. Exact jax top_k tie-breaking (lower index wins on ties).
__global__ __launch_bounds__(256) void router_kernel(
    const float* __restrict__ x, const float* __restrict__ rW, const float* __restrict__ rb,
    float* __restrict__ mu, float* __restrict__ rsig, int* __restrict__ counts,
    int* __restrict__ perm, float* __restrict__ gatew) {
    const int wid  = threadIdx.x >> 5;
    const int lane = threadIdx.x & 31;
    const int t    = blockIdx.x * 8 + wid;

    float sx = 0.f, sxx = 0.f;
    float lg[Ee];
#pragma unroll
    for (int e = 0; e < Ee; ++e) lg[e] = 0.f;

    const float* xr = x + (size_t)t * Hh;
#pragma unroll 4
    for (int i = 0; i < Hh / 32; ++i) {
        const int c = lane + i * 32;
        const float v = xr[c];
        sx += v; sxx += v * v;
        const float* w = rW + (size_t)c * Ee;
#pragma unroll
        for (int e = 0; e < Ee; ++e) lg[e] += v * w[e];
    }
    // wave32 butterfly reductions
#pragma unroll
    for (int off = 16; off >= 1; off >>= 1) {
        sx  += __shfl_xor(sx,  off, 32);
        sxx += __shfl_xor(sxx, off, 32);
#pragma unroll
        for (int e = 0; e < Ee; ++e) lg[e] += __shfl_xor(lg[e], off, 32);
    }
    if (lane == 0) {
        const float m  = sx * (1.0f / Hh);
        const float var = sxx * (1.0f / Hh) - m * m;
        mu[t]   = m;
        rsig[t] = rsqrtf(var + EPS);
#pragma unroll
        for (int e = 0; e < Ee; ++e) lg[e] += rb[e];
        // top-2, strict > scan ascending -> ties pick lower index (jax semantics)
        int i0 = 0; float v0 = lg[0];
#pragma unroll
        for (int e = 1; e < Ee; ++e) if (lg[e] > v0) { v0 = lg[e]; i0 = e; }
        int i1 = -1; float v1 = -3.4e38f;
#pragma unroll
        for (int e = 0; e < Ee; ++e) if (e != i0 && lg[e] > v1) { v1 = lg[e]; i1 = e; }
        // softmax over the two selected logits
        const float e1 = __expf(v1 - v0);
        const float inv = 1.0f / (1.0f + e1);
        const float g0 = inv, g1 = e1 * inv;
        int p0 = atomicAdd(&counts[i0], 1);
        perm[(size_t)i0 * TOK + p0]  = t * 2 + 0;
        gatew[(size_t)i0 * TOK + p0] = g0;
        int p1 = atomicAdd(&counts[i1], 1);
        perm[(size_t)i1 * TOK + p1]  = t * 2 + 1;
        gatew[(size_t)i1 * TOK + p1] = g1;
    }
}

// ---------------- kernel 3: fused grouped-GEMM expert FFN (bf16 WMMA) ----------------
// 1 workgroup = (expert, 32-token tile); 8 waves; dynamic LDS:
//   sh_h  : 32x1024 bf16 (64 KB)   normalized+affine inputs
//   sh_w  : 2 x 64x64 bf16 (16 KB) double-buffered weight staging (transposed [n][k])
//   sh_ac : 32x64  bf16 (4 KB)     GELU activations chunk
static constexpr int SMEM_H  = BM * Hh * 2;             // 65536
static constexpr int SMEM_W1B = 64 * 64 * 2;            // 8192 (one buffer)
static constexpr int SMEM_W  = 2 * SMEM_W1B;            // 16384
static constexpr int SMEM_AC = BM * 64 * 2;             // 4096
static constexpr int SMEM_TS = BM * 4;                  // 128
static constexpr int SMEM_G  = BM * 4;                  // 128
static constexpr int SMEM_TOTAL = SMEM_H + SMEM_W + SMEM_AC + SMEM_TS + SMEM_G;  // 86272

__global__ __launch_bounds__(256, 1) void ffn_kernel(
    const float* __restrict__ x, const float* __restrict__ mu, const float* __restrict__ rsig,
    const float* __restrict__ ln_g, const float* __restrict__ ln_b,
    const __bf16* __restrict__ w1t, const float* __restrict__ b1,
    const __bf16* __restrict__ w2t, const float* __restrict__ b2,
    const int* __restrict__ counts, const int* __restrict__ perm,
    const float* __restrict__ gatew, float* __restrict__ ybuf) {
    extern __shared__ char smem[];
    __bf16* sh_h  = (__bf16*)smem;
    __bf16* sh_wb[2] = { (__bf16*)(smem + SMEM_H), (__bf16*)(smem + SMEM_H + SMEM_W1B) };
    __bf16* sh_ac = (__bf16*)(smem + SMEM_H + SMEM_W);
    int*    sh_ts = (int*)  (smem + SMEM_H + SMEM_W + SMEM_AC);
    float*  sh_g  = (float*)(smem + SMEM_H + SMEM_W + SMEM_AC + SMEM_TS);

    const int e    = blockIdx.x >> 7;           // / MAX_TILES
    const int tile = blockIdx.x & (MAX_TILES - 1);
    const int cnt  = counts[e];
    const int base = tile * BM;
    if (base >= cnt) return;
    const int rows = min(BM, cnt - base);

    const int tid  = threadIdx.x;
    const int lane = tid & 31;
    const int w    = tid >> 5;          // wave 0..7
    const int sub  = lane & 15;
    const int hlf  = lane >> 4;
    const int mi   = w & 1;             // M sub-tile (2 x 16 rows)
    const int nloc = w >> 1;            // N sub-tile within a 64-wide chunk (4 x 16 cols)

    if (tid < BM) {
        if (tid < rows) {
            sh_ts[tid] = perm[(size_t)e * TOK + base + tid];
            sh_g[tid]  = gatew[(size_t)e * TOK + base + tid];
        } else { sh_ts[tid] = 0; sh_g[tid] = 0.f; }
    }
    __syncthreads();

    // ---- build h tile: h = (x - mu) * rsig * ln_g[e] + ln_b[e]  (bf16) ----
    for (int i4 = tid; i4 < BM * (Hh / 4); i4 += 256) {
        const int r  = i4 >> 8;             // / (H/4)
        const int c4 = (i4 & 255) * 4;
        float4 hv = make_float4(0.f, 0.f, 0.f, 0.f);
        if (r < rows) {
            const int t = sh_ts[r] >> 1;
            const float m_ = mu[t], rs = rsig[t];
            const float4 xv = *(const float4*)(x + (size_t)t * Hh + c4);
            const float4 gv = *(const float4*)(ln_g + (size_t)e * Hh + c4);
            const float4 bv = *(const float4*)(ln_b + (size_t)e * Hh + c4);
            hv.x = (xv.x - m_) * rs * gv.x + bv.x;
            hv.y = (xv.y - m_) * rs * gv.y + bv.y;
            hv.z = (xv.z - m_) * rs * gv.z + bv.z;
            hv.w = (xv.w - m_) * rs * gv.w + bv.w;
        }
        __bf16* p = sh_h + r * Hh + c4;
        p[0] = (__bf16)hv.x; p[1] = (__bf16)hv.y; p[2] = (__bf16)hv.z; p[3] = (__bf16)hv.w;
    }
    __syncthreads();

    // output accumulators: wave w owns OUT sub-tiles (mi, c2*4 + nloc) for c2 = 0..15
    v8f acc[16];
#pragma unroll
    for (int i = 0; i < 16; ++i) acc[i] = (v8f){0.f, 0.f, 0.f, 0.f, 0.f, 0.f, 0.f, 0.f};

    const __bf16* w1e = w1t + (size_t)e * FFf * Hh;   // [FF][H]
    const __bf16* w2e = w2t + (size_t)e * Hh * FFf;   // [H][FF]
    const __bf16* arow  = sh_h + (mi * 16 + sub) * Hh + 8 * hlf;
    const __bf16* a2row = sh_ac + (mi * 16 + sub) * 64 + 8 * hlf;
    const int browOff = (nloc * 16 + sub) * 64 + 8 * hlf;

    for (int nblk = 0; nblk < FFf / 64; ++nblk) {
        // ---------- GEMM1: c1[32x64] = h[32x1024] @ W1[:, nblk*64 .. +64) ----------
        v8f c1 = (v8f){0.f, 0.f, 0.f, 0.f, 0.f, 0.f, 0.f, 0.f};
        const __bf16* w1blk = w1e + (size_t)(nblk * 64) * Hh;
        stage64x64(w1blk, Hh, sh_wb[0], tid);           // prime buffer 0
#pragma unroll
        for (int kb = 0; kb < Hh / 64; ++kb) {
            const int cur = kb & 1, nxt = cur ^ 1;
            if (kb + 1 < Hh / 64) {
                stage64x64(w1blk + (kb + 1) * 64, Hh, sh_wb[nxt], tid);
                WAIT_STAGE(2);                          // previous tile landed, next in flight
            } else {
                WAIT_STAGE(0);
            }
            __syncthreads();
            const __bf16* br = sh_wb[cur] + browOff;
#pragma unroll
            for (int kt = 0; kt < 2; ++kt) {
                v16bf a = load_frag(arow, kb * 64 + kt * 32);
                v16bf b = load_frag(br, kt * 32);
                c1 = __builtin_amdgcn_wmma_f32_16x16x32_bf16(
                         false, a, false, b, (short)0, c1, false, false);
            }
            __syncthreads();
        }
        // ---------- bias + exact GELU -> bf16 activation chunk in LDS ----------
        {
            const float bias1 = b1[(size_t)e * FFf + nblk * 64 + nloc * 16 + sub];
#pragma unroll
            for (int r = 0; r < 8; ++r) {
                const int m = mi * 16 + r + 8 * hlf;
                const int n = nloc * 16 + sub;
                const float v = c1[r] + bias1;
                const float g = 0.5f * v * (1.0f + erff(v * 0.70710678118654752f));
                sh_ac[m * 64 + n] = (__bf16)g;
            }
        }
        __syncthreads();
        // ---------- GEMM2 partial: OUT[32x1024] += ac[32x64] @ W2[nblk*64 .. , :] ----------
        const __bf16* w2blk = w2e + nblk * 64;
        stage64x64(w2blk, FFf, sh_wb[0], tid);          // prime buffer 0
#pragma unroll
        for (int c2 = 0; c2 < 16; ++c2) {
            const int cur = c2 & 1, nxt = cur ^ 1;
            if (c2 + 1 < 16) {
                stage64x64(w2blk + (size_t)((c2 + 1) * 64) * FFf, FFf, sh_wb[nxt], tid);
                WAIT_STAGE(2);
            } else {
                WAIT_STAGE(0);
            }
            __syncthreads();
            const __bf16* br = sh_wb[cur] + browOff;
#pragma unroll
            for (int kt = 0; kt < 2; ++kt) {
                v16bf a = load_frag(a2row, kt * 32);
                v16bf b = load_frag(br, kt * 32);
                acc[c2] = __builtin_amdgcn_wmma_f32_16x16x32_bf16(
                              false, a, false, b, (short)0, acc[c2], false, false);
            }
            __syncthreads();
        }
    }

    // ---------- epilogue: ybuf[slot][token][col] = gate * (out + b2) ----------
#pragma unroll
    for (int c2 = 0; c2 < 16; ++c2) {
        const int col = (c2 * 4 + nloc) * 16 + sub;
        const float bias2 = b2[(size_t)e * Hh + col];
#pragma unroll
        for (int r = 0; r < 8; ++r) {
            const int m = mi * 16 + r + 8 * hlf;
            if (m < rows) {
                const int ts = sh_ts[m];
                const int t = ts >> 1, slot = ts & 1;
                ybuf[((size_t)slot * TOK + t) * Hh + col] = sh_g[m] * (acc[c2][r] + bias2);
            }
        }
    }
}

// ---------------- kernel 4: out = x + ybuf0 + ybuf1 (residual; sum(gates)=1) ----------------
__global__ __launch_bounds__(256) void combine_kernel(const float* __restrict__ x,
                                                      const float* __restrict__ yb,
                                                      float* __restrict__ out) {
    const size_t i4 = (size_t)blockIdx.x * 256 + threadIdx.x;
    const float4 a  = ((const float4*)x)[i4];
    const float4 y0 = ((const float4*)yb)[i4];
    const float4 y1 = ((const float4*)(yb + (size_t)TOK * Hh))[i4];
    float4 o;
    o.x = a.x + y0.x + y1.x;
    o.y = a.y + y0.y + y1.y;
    o.z = a.z + y0.z + y1.z;
    o.w = a.w + y0.w + y1.w;
    ((float4*)out)[i4] = o;
}

// ---------------- host launcher ----------------
extern "C" void kernel_launch(void* const* d_in, const int* in_sizes, int n_in,
                              void* d_out, int out_size, void* d_ws, size_t ws_size,
                              hipStream_t stream) {
    (void)in_sizes; (void)n_in; (void)out_size; (void)ws_size;
    const float* x    = (const float*)d_in[0];
    const float* rW   = (const float*)d_in[1];
    const float* rb   = (const float*)d_in[2];
    const float* ln_g = (const float*)d_in[3];
    const float* ln_b = (const float*)d_in[4];
    const float* W1   = (const float*)d_in[5];
    const float* b1   = (const float*)d_in[6];
    const float* W2   = (const float*)d_in[7];
    const float* b2   = (const float*)d_in[8];
    float* out = (float*)d_out;

    char* ws = (char*)d_ws;
    __bf16* w1t  = (__bf16*)(ws + OFF_W1T);
    __bf16* w2t  = (__bf16*)(ws + OFF_W2T);
    float*  ybuf = (float*)(ws + OFF_YB);
    float*  mu   = (float*)(ws + OFF_MU);
    float*  rsig = (float*)(ws + OFF_RS);
    int*    cnts = (int*)(ws + OFF_CNT);
    int*    prm  = (int*)(ws + OFF_PRM);
    float*  gtw  = (float*)(ws + OFF_GTW);

    (void)hipMemsetAsync(cnts, 0, Ee * sizeof(int), stream);

    // W1 [E][H][FF] -> bf16 [E][FF][H];  W2 [E][FF][H] -> bf16 [E][H][FF]
    transpose_cvt_kernel<<<dim3(FFf / 32, Hh / 32, Ee), dim3(32, 8), 0, stream>>>(W1, w1t, Hh, FFf);
    transpose_cvt_kernel<<<dim3(Hh / 32, FFf / 32, Ee), dim3(32, 8), 0, stream>>>(W2, w2t, FFf, Hh);

    router_kernel<<<TOK / 8, 256, 0, stream>>>(x, rW, rb, mu, rsig, cnts, prm, gtw);

    (void)hipFuncSetAttribute((const void*)ffn_kernel,
                              hipFuncAttributeMaxDynamicSharedMemorySize, SMEM_TOTAL);
    ffn_kernel<<<Ee * MAX_TILES, 256, SMEM_TOTAL, stream>>>(
        x, mu, rsig, ln_g, ln_b, w1t, b1, w2t, b2, cnts, prm, gtw, ybuf);

    combine_kernel<<<(TOK * Hh) / (256 * 4), 256, 0, stream>>>(x, ybuf, out);
}
